// EgoGCNeXtC_31430570672675
// MI455X (gfx1250) — matrix operand order repaired and checked by reading, not compiled, not run on hardware
//
#include <hip/hip_runtime.h>

typedef __bf16 bf16;
typedef __attribute__((ext_vector_type(8)))  bf16  v8bf;
typedef __attribute__((ext_vector_type(16))) bf16  v16bf;
typedef __attribute__((ext_vector_type(8)))  float v8f;

#define BB   8
#define CC   128
#define NN   2048
#define KNN  10
#define GG   32
#define NP   11            // N*11 columns for the s-branch
#define BN_RS 0.9999950000374997f   // 1/sqrt(1+1e-5)

static inline int cdiv(int a, int b) { return (a + b - 1) / b; }

// ---------------------------------------------------------------------------
// bf16 WMMA GEMM, 2x2 register-blocked: each wave computes a 32x32 D block
// (4 x v_wmma_f32_16x16x32_bf16 per K-step from 2 A + 2 B fragments).
// A: row-major MxK (A[m*lda+k], K-contiguous)
// B: K-contiguous per column (B[n*ldb+k])  — i.e. B^T row-major
// D: position-major (D[n*ldd+m])
// Block = 4 waves covering 128 consecutive rows (weights stay WGP$-resident).
// MODE: 0=bias  1=bias+relu  2=raw  3=bias+bn+relu
// ---------------------------------------------------------------------------
template <int MODE>
__global__ __launch_bounds__(128) void wmma_gemm(
    const bf16* __restrict__ A, int lda, long long sA,
    const bf16* __restrict__ Bm, int ldb, long long sB,
    float* __restrict__ D, int ldd, long long sD,
    int M, int Nn, int Kk,
    const float* __restrict__ bias,
    const float* __restrict__ bng,
    const float* __restrict__ bnb)
{
  const int lane = threadIdx.x & 31;
  const int wave = threadIdx.x >> 5;
  const int mt = blockIdx.y * 4 + wave;   // 32-row block index
  const int nt = blockIdx.x;              // 32-col block index
  if (mt * 32 >= M) return;
  const int b = blockIdx.z;
  A  += (long long)b * sA;
  Bm += (long long)b * sB;
  D  += (long long)b * sD;

  const int half = lane >> 4;     // 0/1: lane halves per ISA fragment layout
  const int l16  = lane & 15;

  const bf16* arow0 = A  + (size_t)(mt * 32 + l16) * lda;
  const bf16* arow1 = A  + (size_t)(mt * 32 + 16 + l16) * lda;
  const bf16* brow0 = Bm + (size_t)(nt * 32 + l16) * ldb;
  const bf16* brow1 = Bm + (size_t)(nt * 32 + 16 + l16) * ldb;

  v8f acc00 = {}, acc01 = {}, acc10 = {}, acc11 = {};
  for (int k0 = 0; k0 < Kk; k0 += 32) {
    // A 16x32 bf16 fragment: lanes 0-15 -> K {0..7,16..23}, lanes 16-31 -> K {8..15,24..31}
    v8bf a0l = *(const v8bf*)(arow0 + k0 + half * 8);
    v8bf a0h = *(const v8bf*)(arow0 + k0 + 16 + half * 8);
    v8bf a1l = *(const v8bf*)(arow1 + k0 + half * 8);
    v8bf a1h = *(const v8bf*)(arow1 + k0 + 16 + half * 8);
    // B 32x16 bf16 fragment: lanes 0-15 -> K {0..7,16..23}, lanes 16-31 -> K {8..15,24..31}
    v8bf b0l = *(const v8bf*)(brow0 + k0 + half * 8);
    v8bf b0h = *(const v8bf*)(brow0 + k0 + 16 + half * 8);
    v8bf b1l = *(const v8bf*)(brow1 + k0 + half * 8);
    v8bf b1h = *(const v8bf*)(brow1 + k0 + 16 + half * 8);
    v16bf a0 = __builtin_shufflevector(a0l, a0h, 0,1,2,3,4,5,6,7,8,9,10,11,12,13,14,15);
    v16bf a1 = __builtin_shufflevector(a1l, a1h, 0,1,2,3,4,5,6,7,8,9,10,11,12,13,14,15);
    v16bf b0 = __builtin_shufflevector(b0l, b0h, 0,1,2,3,4,5,6,7,8,9,10,11,12,13,14,15);
    v16bf b1 = __builtin_shufflevector(b1l, b1h, 0,1,2,3,4,5,6,7,8,9,10,11,12,13,14,15);
    acc00 = __builtin_amdgcn_wmma_f32_16x16x32_bf16(false, a0, false, b0, (short)0, acc00, false, false);
    acc01 = __builtin_amdgcn_wmma_f32_16x16x32_bf16(false, a0, false, b1, (short)0, acc01, false, false);
    acc10 = __builtin_amdgcn_wmma_f32_16x16x32_bf16(false, a1, false, b0, (short)0, acc10, false, false);
    acc11 = __builtin_amdgcn_wmma_f32_16x16x32_bf16(false, a1, false, b1, (short)0, acc11, false, false);
  }

  // D layout per tile: VGPR v -> M = v + half*8 (contiguous in v), col = l16
  const v8f* accs[4] = {&acc00, &acc01, &acc10, &acc11};
#pragma unroll
  for (int i = 0; i < 2; ++i) {
#pragma unroll
    for (int j = 0; j < 2; ++j) {
      const v8f& acc = *accs[i * 2 + j];
      const int mbase = mt * 32 + i * 16 + half * 8;
      float* dst = D + (size_t)(nt * 32 + j * 16 + l16) * ldd + mbase;
      float r[8];
#pragma unroll
      for (int v = 0; v < 8; ++v) {
        float val = acc[v];
        const int m = mbase + v;
        if constexpr (MODE != 2) val += bias[m];
        if constexpr (MODE == 3) val = val * (bng[m] * BN_RS) + bnb[m];
        if constexpr (MODE == 1 || MODE == 3) val = fmaxf(val, 0.0f);
        r[v] = val;
      }
      *(float4*)(dst + 0) = make_float4(r[0], r[1], r[2], r[3]);
      *(float4*)(dst + 4) = make_float4(r[4], r[5], r[6], r[7]);
    }
  }
}

// ------------------------- helper / VALU kernels ---------------------------

// f32 weight -> bf16 copy
__global__ void k_w2bf(const float* __restrict__ src, bf16* __restrict__ dst, int n) {
  int t = blockIdx.x * blockDim.x + threadIdx.x;
  if (t < n) dst[t] = (bf16)src[t];
}

// Stacked weights for the fmap fusion: rows 0..127 = W1, rows 128..255 = W2-W1
__global__ void k_wst(const float* __restrict__ fmap_w, bf16* __restrict__ wst) {
  int t = blockIdx.x * blockDim.x + threadIdx.x;
  if (t >= 256 * CC) return;
  int c = t & (CC - 1);
  int o = t >> 7;
  float v;
  if (o < CC) v = fmap_w[o * 2 * CC + c];                                  // W1
  else        v = fmap_w[(o - CC) * 2 * CC + CC + c] - fmap_w[(o - CC) * 2 * CC + c]; // W2-W1
  wst[t] = (bf16)v;
}

// x (B,C,N) -> bf16 position-major xbfT[b][n][c]
__global__ void k_xbf(const float* __restrict__ x, bf16* __restrict__ xbfT) {
  int t = blockIdx.x * blockDim.x + threadIdx.x;
  if (t >= BB * NN * CC) return;
  int c = t & (CC - 1);
  int n = (t >> 7) & (NN - 1);
  int b = t >> 18;
  xbfT[t] = (bf16)x[(size_t)b * CC * NN + (size_t)c * NN + n];
}

// xx[b][n] = sum_c x^2
__global__ void k_xx(const float* __restrict__ x, float* __restrict__ xx) {
  int t = blockIdx.x * blockDim.x + threadIdx.x;
  if (t >= BB * NN) return;
  int n = t & (NN - 1);
  int b = t >> 11;
  const float* xp = x + (size_t)b * CC * NN + n;
  float s = 0.f;
  for (int c = 0; c < CC; ++c) { float v = xp[(size_t)c * NN]; s += v * v; }
  xx[t] = s;
}

// topic_mapped[b][c] = relu(intent_w @ topic + intent_b)
__global__ void k_topic(const float* __restrict__ iw, const float* __restrict__ ib,
                        const float* __restrict__ tf, float* __restrict__ out) {
  int t = blockIdx.x * blockDim.x + threadIdx.x;
  if (t >= BB * CC) return;
  int c = t & (CC - 1);
  int b = t >> 7;
  float s = ib[c];
  const float* w = iw + (size_t)c * 256;
  const float* f = tf + (size_t)b * 256;
  for (int j = 0; j < 256; ++j) s += w[j] * f[j];
  out[t] = fmaxf(s, 0.f);
}

// top-10 over score = 2*gram - xx[n] - xx[m]  (stable insertion matches lax.top_k)
__global__ void k_topk(const float* __restrict__ gram, const float* __restrict__ xxb,
                       int* __restrict__ idxb) {
  int n = blockIdx.x * blockDim.x + threadIdx.x;
  if (n >= NN) return;
  const float* row = gram + (size_t)n * NN;
  const float xxn = xxb[n];
  float bv[KNN]; int bi[KNN];
#pragma unroll
  for (int j = 0; j < KNN; ++j) { bv[j] = -INFINITY; bi[j] = 0; }
  for (int m = 0; m < NN; ++m) {
    float s = 2.f * row[m] - xxn - xxb[m];
    if (s > bv[KNN - 1]) {
      int j = KNN - 1;
      while (j > 0 && s > bv[j - 1]) { bv[j] = bv[j - 1]; bi[j] = bi[j - 1]; --j; }
      bv[j] = s; bi[j] = m;
    }
  }
  int* o = idxb + (size_t)n * KNN;
#pragma unroll
  for (int j = 0; j < KNN; ++j) o[j] = bi[j];
}

// grouped conv1d k=3 + bias + relu, reading h1T (pos-major f32), writing bf16
__global__ void k_t2(const float* __restrict__ h1T, const float* __restrict__ w,
                     const float* __restrict__ bias, bf16* __restrict__ h2bf) {
  int t = blockIdx.x * blockDim.x + threadIdx.x;
  if (t >= BB * NN * CC) return;
  int o = t & (CC - 1);
  int n = (t >> 7) & (NN - 1);
  int b = t >> 18;
  int g = o >> 2;
  const float* hb = h1T + (size_t)b * NN * CC;
  float acc = bias[o];
#pragma unroll
  for (int tap = 0; tap < 3; ++tap) {
    int nn = n + tap - 1;
    if (nn < 0 || nn >= NN) continue;
    const float* hp = hb + (size_t)nn * CC + g * 4;
#pragma unroll
    for (int i = 0; i < 4; ++i) acc += w[o * 12 + i * 3 + tap] * hp[i];
  }
  h2bf[t] = (bf16)fmaxf(acc, 0.f);
}

// merged[pos=n*11+k][c] (bf16): k<10 -> relu(P[idx]+Q[n]+fmap_b), k==10 -> topic
__global__ void k_merge(const float* __restrict__ PQ, const int* __restrict__ idxb,
                        const float* __restrict__ topicT, const float* __restrict__ fmap_b,
                        bf16* __restrict__ merged, int b) {
  int t = blockIdx.x * blockDim.x + threadIdx.x;
  if (t >= NN * NP * CC) return;
  int c = t & (CC - 1);
  int p = t >> 7;
  int k = p % NP;
  int n = p / NP;
  float v;
  if (k == NP - 1) {
    v = topicT[b * CC + c];
  } else {
    int nb = idxb[n * KNN + k];
    v = PQ[(size_t)nb * 256 + c] + PQ[(size_t)n * 256 + 128 + c] + fmap_b[c];
    v = fmaxf(v, 0.f);
  }
  merged[t] = (bf16)v;
}

// grouped 1x1 (s2) + bn2 + relu, then mean over the 11 columns -> bf16 smean
__global__ void k_s2mean(const float* __restrict__ sT, const float* __restrict__ w,
                         const float* __restrict__ bias, const float* __restrict__ bng,
                         const float* __restrict__ bnb, bf16* __restrict__ smean, int b) {
  int t = blockIdx.x * blockDim.x + threadIdx.x;
  if (t >= NN * CC) return;
  int o = t & (CC - 1);
  int n = t >> 7;
  int g = o >> 2;
  const float w0 = w[o * 4 + 0], w1 = w[o * 4 + 1], w2 = w[o * 4 + 2], w3 = w[o * 4 + 3];
  const float b0 = bias[o];
  const float sc = bng[o] * BN_RS, bb = bnb[o];
  float acc = 0.f;
#pragma unroll
  for (int k = 0; k < NP; ++k) {
    const float* sp = sT + ((size_t)n * NP + k) * CC + g * 4;
    float v = b0 + w0 * sp[0] + w1 * sp[1] + w2 * sp[2] + w3 * sp[3];
    acc += fmaxf(v * sc + bb, 0.f);
  }
  smean[(size_t)b * NN * CC + t] = (bf16)(acc * (1.f / (float)NP));
}

// out[b,c,n] = relu(tout + x + sout)
__global__ void k_final(const float* __restrict__ toutT, const float* __restrict__ x,
                        const float* __restrict__ soutT, float* __restrict__ out) {
  int t = blockIdx.x * blockDim.x + threadIdx.x;
  if (t >= BB * CC * NN) return;
  int n = t & (NN - 1);
  int c = (t >> 11) & (CC - 1);
  int b = t >> 18;
  size_t pm = (size_t)(b * NN + n) * CC + c;
  out[t] = fmaxf(toutT[pm] + x[t] + soutT[pm], 0.f);
}

// ---------------------------------------------------------------------------

extern "C" void kernel_launch(void* const* d_in, const int* in_sizes, int n_in,
                              void* d_out, int out_size, void* d_ws, size_t ws_size,
                              hipStream_t stream) {
  const float* x        = (const float*)d_in[0];
  const float* topic_f  = (const float*)d_in[1];
  const float* intent_w = (const float*)d_in[2];
  const float* intent_b = (const float*)d_in[3];
  const float* fmap_w   = (const float*)d_in[4];
  const float* fmap_b   = (const float*)d_in[5];
  const float* t1_w     = (const float*)d_in[6];
  const float* t1_b     = (const float*)d_in[7];
  const float* t2_w     = (const float*)d_in[8];
  const float* t2_b     = (const float*)d_in[9];
  const float* t3_w     = (const float*)d_in[10];
  const float* t3_b     = (const float*)d_in[11];
  const float* s1_w     = (const float*)d_in[12];
  const float* s1_b     = (const float*)d_in[13];
  const float* bn1_g    = (const float*)d_in[14];
  const float* bn1_b    = (const float*)d_in[15];
  const float* s2_w     = (const float*)d_in[16];
  const float* s2_b     = (const float*)d_in[17];
  const float* bn2_g    = (const float*)d_in[18];
  const float* bn2_b    = (const float*)d_in[19];
  const float* s3_w     = (const float*)d_in[20];
  const float* s3_b     = (const float*)d_in[21];
  float* out = (float*)d_out;

  // workspace carve-up (256B aligned)
  char* p = (char*)d_ws;
  auto alloc = [&](size_t bytes) -> char* {
    char* r = p;
    p += (bytes + 255) & ~(size_t)255;
    return r;
  };
  bf16*  xbfT   = (bf16*)alloc((size_t)BB * NN * CC * 2);
  float* xx     = (float*)alloc((size_t)BB * NN * 4);
  float* gram   = (float*)alloc((size_t)NN * NN * 4);          // per-batch reuse
  int*   idx    = (int*)alloc((size_t)BB * NN * KNN * 4);
  bf16*  wt1    = (bf16*)alloc((size_t)CC * CC * 2);
  bf16*  wt3    = (bf16*)alloc((size_t)CC * CC * 2);
  bf16*  ws1    = (bf16*)alloc((size_t)CC * CC * 2);
  bf16*  ws3    = (bf16*)alloc((size_t)CC * CC * 2);
  bf16*  wst    = (bf16*)alloc((size_t)256 * CC * 2);
  float* h1T    = (float*)alloc((size_t)BB * NN * CC * 4);
  bf16*  h2bf   = (bf16*)alloc((size_t)BB * NN * CC * 2);
  float* toutT  = (float*)alloc((size_t)BB * NN * CC * 4);
  float* PQT    = (float*)alloc((size_t)BB * NN * 256 * 4);
  float* topicT = (float*)alloc((size_t)BB * CC * 4);
  bf16*  merged = (bf16*)alloc((size_t)NN * NP * CC * 2);      // per-batch reuse
  float* sT     = (float*)alloc((size_t)NN * NP * CC * 4);     // per-batch reuse
  bf16*  smean  = (bf16*)alloc((size_t)BB * NN * CC * 2);
  float* soutT  = (float*)alloc((size_t)BB * NN * CC * 4);

  const int TB = 256;
  const int GB = 128;   // GEMM block = 4 waves (covers 128 rows)

  // weights -> bf16
  k_w2bf<<<cdiv(CC * CC, TB), TB, 0, stream>>>(t1_w, wt1, CC * CC);
  k_w2bf<<<cdiv(CC * CC, TB), TB, 0, stream>>>(t3_w, wt3, CC * CC);
  k_w2bf<<<cdiv(CC * CC, TB), TB, 0, stream>>>(s1_w, ws1, CC * CC);
  k_w2bf<<<cdiv(CC * CC, TB), TB, 0, stream>>>(s3_w, ws3, CC * CC);
  k_wst<<<cdiv(256 * CC, TB), TB, 0, stream>>>(fmap_w, wst);

  // activation prep
  k_xbf<<<cdiv(BB * NN * CC, TB), TB, 0, stream>>>(x, xbfT);
  k_xx<<<cdiv(BB * NN, TB), TB, 0, stream>>>(x, xx);
  k_topic<<<cdiv(BB * CC, TB), TB, 0, stream>>>(intent_w, intent_b, topic_f, topicT);

  const long long sAct = (long long)NN * CC;   // per-batch activation stride

  // t-branch: h1 = relu(t1 @ x)
  wmma_gemm<1><<<dim3(NN / 32, 1, BB), GB, 0, stream>>>(
      wt1, CC, 0, xbfT, CC, sAct, h1T, CC, sAct, CC, NN, CC, t1_b, nullptr, nullptr);
  k_t2<<<cdiv(BB * NN * CC, TB), TB, 0, stream>>>(h1T, t2_w, t2_b, h2bf);
  // tout = t3 @ h2
  wmma_gemm<0><<<dim3(NN / 32, 1, BB), GB, 0, stream>>>(
      wt3, CC, 0, h2bf, CC, sAct, toutT, CC, sAct, CC, NN, CC, t3_b, nullptr, nullptr);

  // PQ = [W1 ; W2-W1] @ x  (fmap fusion)
  wmma_gemm<2><<<dim3(NN / 32, 2, BB), GB, 0, stream>>>(
      wst, CC, 0, xbfT, CC, sAct, PQT, 256, (long long)NN * 256, 256, NN, CC,
      nullptr, nullptr, nullptr);

  // per-batch: Gram (WMMA) + top-k, then s-branch
  for (int b = 0; b < BB; ++b) {
    const bf16* xb = xbfT + (size_t)b * NN * CC;
    wmma_gemm<2><<<dim3(NN / 32, NN / 128, 1), GB, 0, stream>>>(
        xb, CC, 0, xb, CC, 0, gram, NN, 0, NN, NN, CC, nullptr, nullptr, nullptr);
    k_topk<<<cdiv(NN, 128), 128, 0, stream>>>(gram, xx + (size_t)b * NN,
                                              idx + (size_t)b * NN * KNN);

    k_merge<<<cdiv(NN * NP * CC, TB), TB, 0, stream>>>(
        PQT + (size_t)b * NN * 256, idx + (size_t)b * NN * KNN, topicT, fmap_b, merged, b);
    // s = relu(bn1(s1 @ merged + s1_b))
    wmma_gemm<3><<<dim3(NN * NP / 32, 1, 1), GB, 0, stream>>>(
        ws1, CC, 0, merged, CC, 0, sT, CC, 0, CC, NN * NP, CC, s1_b, bn1_g, bn1_b);
    k_s2mean<<<cdiv(NN * CC, TB), TB, 0, stream>>>(sT, s2_w, s2_b, bn2_g, bn2_b, smean, b);
  }

  // sout = s3 @ smean + s3_b
  wmma_gemm<0><<<dim3(NN / 32, 1, BB), GB, 0, stream>>>(
      ws3, CC, 0, smean, CC, sAct, soutT, CC, sAct, CC, NN, CC, s3_b, nullptr, nullptr);

  // out = relu(tout + x + sout)
  k_final<<<cdiv(BB * CC * NN, TB), TB, 0, stream>>>(toutT, x, soutT, out);
}
